// SelfAttention_7258494730540
// MI455X (gfx1250) — matrix operand order repaired and checked by reading, not compiled, not run on hardware
//
#include <hip/hip_runtime.h>
#include <hip/hip_bf16.h>

// ---------------------------------------------------------------------------
// Causal self-attention for B=4, S=2048, E=A=1024 (fp32 in/out) on gfx1250.
// All GEMMs run on v_wmma_f32_16x16x32_bf16 (fp32 accumulate), with register
// blocking so each fetched fragment feeds multiple WMMAs.
// ---------------------------------------------------------------------------

#define B_   4
#define S_   2048
#define E_   1024
#define A_   1024
#define MTOT (B_ * S_)          // 8192 rows of X/Q/K/V

typedef __attribute__((ext_vector_type(16))) __bf16 bf16x16;
typedef __attribute__((ext_vector_type(8)))  __bf16 bf16x8;
typedef __attribute__((ext_vector_type(8)))  float  f32x8;

// float -> bf16 (round to nearest even, ignoring NaN edge cases)
static __device__ __forceinline__ unsigned short f2bf(float f) {
    unsigned int u = __float_as_uint(f);
    u += 0x7FFFu + ((u >> 16) & 1u);
    return (unsigned short)(u >> 16);
}

// 16x32 bf16 A-fragment per ISA layout: lane = (row = l&15, khalf = l>>4);
// per lane two contiguous runs of 8 bf16 at [k0+khalf*8) and [k0+16+khalf*8).
static __device__ __forceinline__ bf16x16 load_a_frag(const unsigned short* rowPtr,
                                                      int k0, int khalf) {
    const bf16x8 lo = *reinterpret_cast<const bf16x8*>(rowPtr + k0 + khalf * 8);
    const bf16x8 hi = *reinterpret_cast<const bf16x8*>(rowPtr + k0 + 16 + khalf * 8);
    bf16x16 a;
#pragma unroll
    for (int j = 0; j < 8; ++j) { a[j] = lo[j]; a[j + 8] = hi[j]; }
    return a;
}

// 32x16 bf16 B-fragment: lane = (col n = l&15, khalf = l>>4); two runs of 8
// bf16 along K at [k0+khalf*8) and [k0+16+khalf*8) within the column's K-major row.
static __device__ __forceinline__ bf16x16 load_b_frag(const unsigned short* colPtr,
                                                      int k0, int khalf) {
    const bf16x8 lo = *reinterpret_cast<const bf16x8*>(colPtr + k0 + khalf * 8);
    const bf16x8 hi = *reinterpret_cast<const bf16x8*>(colPtr + k0 + 16 + khalf * 8);
    bf16x16 b;
#pragma unroll
    for (int j = 0; j < 8; ++j) { b[j] = lo[j]; b[j + 8] = hi[j]; }
    return b;
}

// ---------------------------------------------------------------------------
// Kernel 1: fp32 -> bf16 bulk convert (n divisible by 1024)
// ---------------------------------------------------------------------------
__global__ void cvt_f32_bf16(const float* __restrict__ src,
                             unsigned short* __restrict__ dst, int n4) {
    int i = blockIdx.x * blockDim.x + threadIdx.x;
    if (i < n4) {
        float4 v = reinterpret_cast<const float4*>(src)[i];
        ushort4 o;
        o.x = f2bf(v.x); o.y = f2bf(v.y); o.z = f2bf(v.z); o.w = f2bf(v.w);
        reinterpret_cast<ushort4*>(dst)[i] = o;
    }
}

// ---------------------------------------------------------------------------
// Kernel 2: QKV projection, 2x4 register-blocked.
// grid = (MTOT/64, A/256, 3); block = 256 (8 waves as 2 (M) x 4 (N)).
// Each wave computes a 32x64 output block: 2 m-tiles x 4 n-tiles of 16x16.
// Per K-step: 2 A-frags + 4 B-frags -> 8 WMMAs.
// kind 0/1 -> Q/K stored bf16 row-major; kind 2 -> V stored bf16 transposed.
// ---------------------------------------------------------------------------
__global__ void qkv_gemm(const unsigned short* __restrict__ Xb,
                         const unsigned short* __restrict__ Wqb,
                         const unsigned short* __restrict__ Wkb,
                         const unsigned short* __restrict__ Wvb,
                         const float* __restrict__ bq,
                         const float* __restrict__ bk,
                         const float* __restrict__ bv,
                         unsigned short* __restrict__ Qb,
                         unsigned short* __restrict__ Kb,
                         unsigned short* __restrict__ Vt) {
    const int lane  = threadIdx.x & 31;
    const int wave  = threadIdx.x >> 5;
    const int nl    = lane & 15;
    const int khalf = lane >> 4;
    const int wm    = wave >> 2;                 // 0..1
    const int wn    = wave & 3;                  // 0..3
    const int kind  = blockIdx.z;

    const int m0 = blockIdx.x * 64 + wm * 32;    // this wave's 32 rows
    const int n0 = blockIdx.y * 256 + wn * 64;   // this wave's 64 cols

    const unsigned short* W   = (kind == 0) ? Wqb : (kind == 1) ? Wkb : Wvb;
    const float*          bia = (kind == 0) ? bq  : (kind == 1) ? bk  : bv;

    const unsigned short* aRow[2];
    aRow[0] = Xb + (size_t)(m0 + nl) * E_;
    aRow[1] = Xb + (size_t)(m0 + 16 + nl) * E_;
    const unsigned short* bCol[4];
#pragma unroll
    for (int j = 0; j < 4; ++j)
        bCol[j] = W + (size_t)(n0 + j * 16 + nl) * E_;

    f32x8 acc[2][4] = {};
#pragma unroll 2
    for (int k0 = 0; k0 < E_; k0 += 32) {
        __builtin_prefetch(bCol[0] + k0 + 128, 0, 1);   // global_prefetch_b8
        bf16x16 a0 = load_a_frag(aRow[0], k0, khalf);
        bf16x16 a1 = load_a_frag(aRow[1], k0, khalf);
#pragma unroll
        for (int j = 0; j < 4; ++j) {
            bf16x16 b = load_b_frag(bCol[j], k0, khalf);
            acc[0][j] = __builtin_amdgcn_wmma_f32_16x16x32_bf16(
                false, a0, false, b, (short)0, acc[0][j], false, false);
            acc[1][j] = __builtin_amdgcn_wmma_f32_16x16x32_bf16(
                false, a1, false, b, (short)0, acc[1][j], false, false);
        }
    }

#pragma unroll
    for (int i = 0; i < 2; ++i) {
#pragma unroll
        for (int j = 0; j < 4; ++j) {
            const int n  = n0 + j * 16 + nl;
            const float bn = bia[n];
            if (kind < 2) {
                unsigned short* dst = (kind == 0) ? Qb : Kb;
#pragma unroll
                for (int r = 0; r < 8; ++r) {    // C: VGPR r -> row m = r + khalf*8
                    const int m = m0 + i * 16 + r + khalf * 8;
                    dst[(size_t)m * A_ + n] = f2bf(acc[i][j][r] + bn);
                }
            } else {
                // Vt[b][a][s]: lane's 8 values are contiguous in s
                const int mbase = m0 + i * 16;
                const int bb = mbase / S_;
                const int s0 = (mbase % S_) + khalf * 8;
                ushort4* dst = reinterpret_cast<ushort4*>(
                    Vt + ((size_t)bb * A_ + n) * S_ + s0);
                ushort4 p0, p1;
                p0.x = f2bf(acc[i][j][0] + bn); p0.y = f2bf(acc[i][j][1] + bn);
                p0.z = f2bf(acc[i][j][2] + bn); p0.w = f2bf(acc[i][j][3] + bn);
                p1.x = f2bf(acc[i][j][4] + bn); p1.y = f2bf(acc[i][j][5] + bn);
                p1.z = f2bf(acc[i][j][6] + bn); p1.w = f2bf(acc[i][j][7] + bn);
                dst[0] = p0; dst[1] = p1;
            }
        }
    }
}

// ---------------------------------------------------------------------------
// Kernel 3: fused causal attention for one 16-row q-tile per workgroup.
// grid = B*(S/16) = 512 blocks; block = 256 threads (8 waves).
// Dynamic LDS: 16*2048 fp32 scores (128KB) + 16*2048 bf16 probs (64KB) = 192KB
// (valid on CDNA5: 320KB LDS per WGP).
// ---------------------------------------------------------------------------
__global__ void attn_fused(const unsigned short* __restrict__ Qb,
                           const unsigned short* __restrict__ Kb,
                           const unsigned short* __restrict__ Vt,
                           float* __restrict__ out) {
    extern __shared__ char smem[];
    float*          scoreS = reinterpret_cast<float*>(smem);                        // [16][2048]
    unsigned short* attnS  = reinterpret_cast<unsigned short*>(smem + 16 * S_ * 4); // [16][2048]

    const int lane  = threadIdx.x & 31;
    const int wave  = threadIdx.x >> 5;
    const int nl    = lane & 15;
    const int khalf = lane >> 4;

    const int qt    = blockIdx.x & (S_ / 16 - 1);
    const int bb    = blockIdx.x >> 7;          // S_/16 == 128
    const int q0    = qt * 16;
    const int Tk    = qt + 1;                   // causal k-tiles
    const int Kcols = Tk * 16;
    const float rscale = 0.03125f;              // 1/sqrt(1024)

    // ---- Phase A: S = Q K^T / sqrt(dk), causal mask, into LDS -------------
    // Pairs of k-tiles per wave so the Q fragment is loaded once per 2 WMMAs.
    const unsigned short* qRow = Qb + ((size_t)(bb * S_ + q0 + nl)) * A_;
    for (int kt0 = wave * 2; kt0 < Tk; kt0 += 16) {
        const int  kt1    = kt0 + 1;
        const bool has1   = (kt1 < Tk);
        const unsigned short* kCol0 = Kb + ((size_t)(bb * S_ + kt0 * 16 + nl)) * A_;
        const unsigned short* kCol1 = Kb + ((size_t)(bb * S_ + kt1 * 16 + nl)) * A_;
        f32x8 acc0 = {}, acc1 = {};
#pragma unroll 4
        for (int k0 = 0; k0 < A_; k0 += 32) {
            bf16x16 a = load_a_frag(qRow, k0, khalf);
            bf16x16 b0 = load_b_frag(kCol0, k0, khalf);
            acc0 = __builtin_amdgcn_wmma_f32_16x16x32_bf16(
                false, a, false, b0, (short)0, acc0, false, false);
            if (has1) {
                bf16x16 b1 = load_b_frag(kCol1, k0, khalf);
                acc1 = __builtin_amdgcn_wmma_f32_16x16x32_bf16(
                    false, a, false, b1, (short)0, acc1, false, false);
            }
        }
        const int kg0 = kt0 * 16 + nl;
        const int kg1 = kt1 * 16 + nl;
#pragma unroll
        for (int r = 0; r < 8; ++r) {
            const int ml = r + khalf * 8;
            float v0 = acc0[r] * rscale;
            if (kg0 > q0 + ml) v0 = -3.0e30f;   // causal mask
            scoreS[ml * S_ + kg0] = v0;
            if (has1) {
                float v1 = acc1[r] * rscale;
                if (kg1 > q0 + ml) v1 = -3.0e30f;
                scoreS[ml * S_ + kg1] = v1;
            }
        }
    }
    __syncthreads();

    // ---- Phase B: row softmax in LDS, emit bf16 probabilities -------------
    {
        const int row = threadIdx.x >> 4;       // 16 threads per row
        const int c   = threadIdx.x & 15;
        float mx = -3.4e38f;
        for (int col = c; col < Kcols; col += 16)
            mx = fmaxf(mx, scoreS[row * S_ + col]);
#pragma unroll
        for (int off = 8; off >= 1; off >>= 1)
            mx = fmaxf(mx, __shfl_xor(mx, off, 32));
        float sum = 0.0f;
        for (int col = c; col < Kcols; col += 16)
            sum += __expf(scoreS[row * S_ + col] - mx);
#pragma unroll
        for (int off = 8; off >= 1; off >>= 1)
            sum += __shfl_xor(sum, off, 32);
        const float inv = 1.0f / sum;
        for (int col = c; col < Kcols; col += 16)
            attnS[row * S_ + col] = f2bf(__expf(scoreS[row * S_ + col] - mx) * inv);
        // zero-pad probabilities up to a multiple of 32 K-columns for WMMA
        const int KcolsPad = (Kcols + 31) & ~31;
        if (Kcols + c < KcolsPad)
            attnS[row * S_ + Kcols + c] = 0;
    }
    __syncthreads();

    // ---- Phase C: O = P @ V, 4-way n-tile blocked ------------------------
    // One LDS A-fragment read feeds 4 WMMAs against 4 Vt columns.
    const int KcolsPad = (Kcols + 31) & ~31;
    const unsigned short* aRow = attnS + (size_t)nl * S_;   // P row = lane&15
#pragma unroll
    for (int g = 0; g < 2; ++g) {
        const int ntBase = wave * 8 + g * 4;
        const unsigned short* vCol[4];
#pragma unroll
        for (int j = 0; j < 4; ++j)
            vCol[j] = Vt + ((size_t)bb * A_ + (ntBase + j) * 16 + nl) * S_;
        f32x8 acc[4] = {};
        for (int k0 = 0; k0 < KcolsPad; k0 += 32) {
            bf16x16 a = load_a_frag(aRow, k0, khalf);       // ds_load from LDS
#pragma unroll
            for (int j = 0; j < 4; ++j) {
                bf16x16 b = load_b_frag(vCol[j], k0, khalf);
                acc[j] = __builtin_amdgcn_wmma_f32_16x16x32_bf16(
                    false, a, false, b, (short)0, acc[j], false, false);
            }
        }
#pragma unroll
        for (int j = 0; j < 4; ++j) {
            const int n = (ntBase + j) * 16 + nl;
#pragma unroll
            for (int r = 0; r < 8; ++r) {
                const int q = q0 + r + khalf * 8;
                out[((size_t)(bb * S_ + q)) * A_ + n] = acc[j][r];
            }
        }
    }
}

// ---------------------------------------------------------------------------
// Host launcher.  Workspace layout (bytes):
//   Xb   bf16 [8192][1024]      16,777,216
//   Wqb/Wkb/Wvb bf16 [1024][1024]  3 * 2,097,152
//   Qb   bf16 [8192][1024]      16,777,216
//   Kb   bf16 [8192][1024]      16,777,216
//   Vt   bf16 [4][1024][2048]   16,777,216        total ~73.4 MB
// ---------------------------------------------------------------------------
extern "C" void kernel_launch(void* const* d_in, const int* in_sizes, int n_in,
                              void* d_out, int out_size, void* d_ws, size_t ws_size,
                              hipStream_t stream) {
    const float* emb = (const float*)d_in[0];
    const float* Wq  = (const float*)d_in[1];
    const float* bq  = (const float*)d_in[2];
    const float* Wk  = (const float*)d_in[3];
    const float* bk  = (const float*)d_in[4];
    const float* Wv  = (const float*)d_in[5];
    const float* bv  = (const float*)d_in[6];
    float* out = (float*)d_out;

    char* ws = (char*)d_ws;
    size_t off = 0;
    unsigned short* Xb  = (unsigned short*)(ws + off); off += (size_t)MTOT * E_ * 2;
    unsigned short* Wqb = (unsigned short*)(ws + off); off += (size_t)A_ * E_ * 2;
    unsigned short* Wkb = (unsigned short*)(ws + off); off += (size_t)A_ * E_ * 2;
    unsigned short* Wvb = (unsigned short*)(ws + off); off += (size_t)A_ * E_ * 2;
    unsigned short* Qb  = (unsigned short*)(ws + off); off += (size_t)MTOT * A_ * 2;
    unsigned short* Kb  = (unsigned short*)(ws + off); off += (size_t)MTOT * A_ * 2;
    unsigned short* Vt  = (unsigned short*)(ws + off); off += (size_t)MTOT * A_ * 2;

    // 1) fp32 -> bf16 staging
    cvt_f32_bf16<<<(MTOT * E_ / 4 + 255) / 256, 256, 0, stream>>>(emb, Xb, MTOT * E_ / 4);
    cvt_f32_bf16<<<(A_ * E_ / 4 + 255) / 256, 256, 0, stream>>>(Wq, Wqb, A_ * E_ / 4);
    cvt_f32_bf16<<<(A_ * E_ / 4 + 255) / 256, 256, 0, stream>>>(Wk, Wkb, A_ * E_ / 4);
    cvt_f32_bf16<<<(A_ * E_ / 4 + 255) / 256, 256, 0, stream>>>(Wv, Wvb, A_ * E_ / 4);

    // 2) Q/K/V projections (grid.z selects the projection)
    qkv_gemm<<<dim3(MTOT / 64, A_ / 256, 3), 256, 0, stream>>>(
        Xb, Wqb, Wkb, Wvb, bq, bk, bv, Qb, Kb, Vt);

    // 3) fused causal attention; 192KB dynamic LDS per workgroup (CDNA5 WGP)
    const size_t ldsBytes = (size_t)16 * S_ * 4 + (size_t)16 * S_ * 2;
    attn_fused<<<B_ * (S_ / 16), 256, ldsBytes, stream>>>(Qb, Kb, Vt, out);
}
